// CNet_77025943487014
// MI455X (gfx1250) — matrix operand order repaired and checked by reading, not compiled, not run on hardware
//
#include <hip/hip_runtime.h>

typedef __attribute__((ext_vector_type(16))) _Float16 v16h;
typedef __attribute__((ext_vector_type(8)))  _Float16 v8h;
typedef __attribute__((ext_vector_type(2)))  _Float16 h2;
typedef __attribute__((ext_vector_type(8)))  float    v8f;
typedef __attribute__((ext_vector_type(4)))  float    f4;

#define B_N   8192
#define NA_N  50
#define NC_N  5
#define H_N   32
#define SCALE 0.17677669529663687f   // 1/sqrt(32)

__device__ inline v8f zero8() {
    v8f z;
#pragma unroll
    for (int i = 0; i < 8; ++i) z[i] = 0.f;
    return z;
}

__device__ inline v8f wmma16(v16h a, v16h b, v8f c) {
    // D = A(16x32 f16) x B(32x16 f16) + C(16x16 f32)
    return __builtin_amdgcn_wmma_f32_16x16x32_f16(false, a, false, b, (short)0, c, false, false);
}

// broadcast from a wave-uniform lane: v_readlane_b32 -> SGPR scalar operand
__device__ inline float rlane(float v, int l) {
    return __uint_as_float(__builtin_amdgcn_readlane(__float_as_uint(v), l));
}

// A-fragment (16x32 f16) straight from a global f16 [rows][32] matrix (16B aligned).
__device__ inline v16h afrag_g32(const _Float16* P, int row, int kh) {
    v8h lo = *(const v8h*)(P + (size_t)row * 32 + kh * 8);
    v8h hi = *(const v8h*)(P + (size_t)row * 32 + 16 + kh * 8);
    v16h a;
#pragma unroll
    for (int j = 0; j < 8; ++j) { a[j] = lo[j]; a[j + 8] = hi[j]; }
    return a;
}

// Pre-pack a [Krows x 32] f32 weight into WMMA B-fragments (f16) in LDS.
// Element i = ((s*2+t)*32+lane)*8+v holds W[k0..k0+1][n] with
//   n = t*16 + (lane&15),  k0 = s*32 + (v>>2)*16 + (lane>>4)*8 + 2*(v&3) ; k>=Krows -> 0 (zero-pad!)
__device__ inline void build_bfrag(h2* dst, const float* W, int Krows, int ksteps,
                                   int tid, int nth) {
    int total = ksteps * 2 * 32 * 8;
    for (int i = tid; i < total; i += nth) {
        int v    = i & 7;
        int lane = (i >> 3) & 31;
        int t    = (i >> 8) & 1;
        int s    = i >> 9;
        int n  = t * 16 + (lane & 15);
        int k0 = s * 32 + (v >> 2) * 16 + (lane >> 4) * 8 + 2 * (v & 3);
        float a0 = (k0     < Krows) ? W[(size_t)k0 * 32 + n]       : 0.f;
        float a1 = (k0 + 1 < Krows) ? W[(size_t)(k0 + 1) * 32 + n] : 0.f;
        h2 hv;
        hv[0] = (_Float16)a0;
        hv[1] = (_Float16)a1;
        dst[i] = hv;
    }
}

__device__ inline void store_tile_f16(_Float16* out, int row0, int nl, int kh,
                                      v8f o0, v8f o1) {
#pragma unroll
    for (int r = 0; r < 8; ++r) {
        int m = r + 8 * kh;
        out[(size_t)(row0 + m) * 32 + nl]      = (_Float16)o0[r];
        out[(size_t)(row0 + m) * 32 + nl + 16] = (_Float16)o1[r];
    }
}

// Bounce a 16x16(+16x16) f32 C/D tile through LDS and re-read as an A fragment.
__device__ inline v16h cd_to_afrag(_Float16* hl, int nl, int kh, v8f c0, v8f c1) {
#pragma unroll
    for (int r = 0; r < 8; ++r) {
        int m = r + 8 * kh;
        hl[m * 32 + nl]      = (_Float16)c0[r];
        hl[m * 32 + nl + 16] = (_Float16)c1[r];
    }
    v8h lo = *(const v8h*)&hl[nl * 32 + kh * 8];
    v8h hi = *(const v8h*)&hl[nl * 32 + 16 + kh * 8];
    v16h a;
#pragma unroll
    for (int j = 0; j < 8; ++j) { a[j] = lo[j]; a[j + 8] = hi[j]; }
    return a;
}

struct EncArgs {
    const float* X;        // [R, KIN] f32
    int R;
    const float* W1; const float* b1;   // [KIN,32], [32]
    const float* W2; const float* b2;   // [32,32],  [32]
    const float* PW[8];                 // projection weights [32,32]
    _Float16*    Pout[8];               // projection outputs [R,32] f16
};

// Fused: h = relu(X@W1+b1)@W2+b2 ; Pout[p] = h @ PW[p]   (all via WMMA)
template <int KIN, int KSTEPS, int NPROJ>
__global__ __launch_bounds__(256) void enc_proj_kernel(EncArgs A) {
    constexpr int KPAD = KSTEPS * 32;
    __shared__ __align__(16) h2 w1f[KSTEPS * 2 * 32 * 8];
    __shared__ __align__(16) h2 w2f[2 * 32 * 8];
    __shared__ __align__(16) h2 pjf[NPROJ][2 * 32 * 8];
    __shared__ __align__(16) _Float16 xtile[8][16 * KPAD]; // per-wave staged A tile
    __shared__ __align__(16) _Float16 hlds[8][16 * 32];    // per-wave bounce tile

    int tid = threadIdx.x;
    build_bfrag(w1f, A.W1, KIN, KSTEPS, tid, 256);
    build_bfrag(w2f, A.W2, 32, 1, tid, 256);
#pragma unroll
    for (int p = 0; p < NPROJ; ++p) build_bfrag(pjf[p], A.PW[p], 32, 1, tid, 256);
    __syncthreads();

    int lane = tid & 31, w = tid >> 5;
    int nl = lane & 15, kh = lane >> 4;
    int ntiles = A.R >> 4;
    int wavesTotal = gridDim.x * 8;

    for (int tile = blockIdx.x * 8 + w; tile < ntiles; tile += wavesTotal) {
        int row0 = tile << 4;

        // stage 16xKIN f32 tile -> f16 LDS (flat contiguous b128 loads).
        // A-side K-padding stays stale: corresponding B rows are zero.
        {
            const float* Xt = A.X + (size_t)row0 * KIN;   // 16B aligned
#pragma unroll
            for (int c0 = 0; c0 < 4 * KIN; c0 += 32) {
                int c = c0 + lane;
                if (c < 4 * KIN) {
                    f4 xv = *(const f4*)(Xt + 4 * c);
#pragma unroll
                    for (int j = 0; j < 4; ++j) {
                        int p = 4 * c + j;
                        int r = p / KIN;
                        int cc = p - r * KIN;
                        xtile[w][r * KPAD + cc] = (_Float16)xv[j];
                    }
                }
            }
        }

        // GEMM1: X @ W1
        v8f acc0 = zero8(), acc1 = zero8();
#pragma unroll
        for (int s = 0; s < KSTEPS; ++s) {
            v8h lo = *(const v8h*)&xtile[w][nl * KPAD + s * 32 + kh * 8];
            v8h hi = *(const v8h*)&xtile[w][nl * KPAD + s * 32 + 16 + kh * 8];
            v16h a;
#pragma unroll
            for (int j = 0; j < 8; ++j) { a[j] = lo[j]; a[j + 8] = hi[j]; }
            v16h b0 = *(const v16h*)&w1f[((s * 2 + 0) * 32 + lane) * 8];
            v16h b1v= *(const v16h*)&w1f[((s * 2 + 1) * 32 + lane) * 8];
            acc0 = wmma16(a, b0, acc0);
            acc1 = wmma16(a, b1v, acc1);
        }
        float bb0 = A.b1[nl], bb1 = A.b1[nl + 16];
#pragma unroll
        for (int r = 0; r < 8; ++r) {
            acc0[r] = fmaxf(acc0[r] + bb0, 0.f);
            acc1[r] = fmaxf(acc1[r] + bb1, 0.f);
        }
        v16h ah = cd_to_afrag(hlds[w], nl, kh, acc0, acc1);

        // GEMM2: h1 @ W2 + b2
        v8f h20 = wmma16(ah, *(const v16h*)&w2f[(0 * 32 + lane) * 8], zero8());
        v8f h21 = wmma16(ah, *(const v16h*)&w2f[(1 * 32 + lane) * 8], zero8());
        bb0 = A.b2[nl]; bb1 = A.b2[nl + 16];
#pragma unroll
        for (int r = 0; r < 8; ++r) { h20[r] += bb0; h21[r] += bb1; }
        v16h ah2 = cd_to_afrag(hlds[w], nl, kh, h20, h21);

        // projections: h2 @ PW[p]
#pragma unroll
        for (int p = 0; p < NPROJ; ++p) {
            v8f o0 = wmma16(ah2, *(const v16h*)&pjf[p][(0 * 32 + lane) * 8], zero8());
            v8f o1 = wmma16(ah2, *(const v16h*)&pjf[p][(1 * 32 + lane) * 8], zero8());
            store_tile_f16(A.Pout[p], row0, nl, kh, o0, o1);
        }
    }
}

// One wave32 per batch element; lane = feature channel (H=32).
// Computes softmax attention per head and stores att (pre-Wo) as f16 planes.
__global__ __launch_bounds__(256) void attn_core_kernel(
    const _Float16* K0, const _Float16* V0,
    const _Float16* K1, const _Float16* V1,
    const _Float16* Qw,                        // 4 planes of [B,5,32]
    const _Float16* K23, const _Float16* V23,  // 2 planes each of [B,5,32]
    _Float16* attw,                            // 4 planes of [B,5,32] (out)
    int Bn) {
    int tid = threadIdx.x, lane = tid & 31, w = tid >> 5;
    int b = blockIdx.x * 8 + w;
    if (b >= Bn) return;

#pragma unroll
    for (int head = 0; head < 4; ++head) {
        const _Float16* Qp = Qw + (size_t)head * Bn * 160 + (size_t)b * 160;
        float qv[5];
#pragma unroll
        for (int q = 0; q < 5; ++q) qv[q] = (float)Qp[q * 32 + lane];

        float att[5] = {0.f, 0.f, 0.f, 0.f, 0.f};

        if (head < 2) {
            const _Float16* Kh = (head == 0) ? K0 : K1;
            const _Float16* Vh = (head == 0) ? V0 : V1;
            float sc0[5], sc1[5];
            {   // keys g = lane (always < 50)
                const h2* kp = (const h2*)(Kh + (size_t)b * 1600 + (size_t)lane * 32);
                h2 kr[16];
#pragma unroll
                for (int j = 0; j < 16; ++j) kr[j] = kp[j];
#pragma unroll
                for (int q = 0; q < 5; ++q) {
                    float s = 0.f;
#pragma unroll
                    for (int j = 0; j < 16; ++j) {
                        s += rlane(qv[q], 2 * j)     * (float)kr[j][0];
                        s += rlane(qv[q], 2 * j + 1) * (float)kr[j][1];
                    }
                    sc0[q] = s * SCALE;
                }
            }
            {   // keys g = lane+32, valid iff g < 50
                int g = lane + 32;
                bool valid = g < 50;
                int ge = valid ? g : 0;
                const h2* kp = (const h2*)(Kh + (size_t)b * 1600 + (size_t)ge * 32);
                h2 kr[16];
#pragma unroll
                for (int j = 0; j < 16; ++j) kr[j] = kp[j];
#pragma unroll
                for (int q = 0; q < 5; ++q) {
                    float s = 0.f;
#pragma unroll
                    for (int j = 0; j < 16; ++j) {
                        s += rlane(qv[q], 2 * j)     * (float)kr[j][0];
                        s += rlane(qv[q], 2 * j + 1) * (float)kr[j][1];
                    }
                    sc1[q] = valid ? s * SCALE : -1e30f;
                }
            }
            float p0[5], p1[5];
#pragma unroll
            for (int q = 0; q < 5; ++q) {
                float m = fmaxf(sc0[q], sc1[q]);
#pragma unroll
                for (int off = 16; off >= 1; off >>= 1) m = fmaxf(m, __shfl_xor(m, off));
                float e0 = __expf(sc0[q] - m);
                float e1 = __expf(sc1[q] - m);
                float ssum = e0 + e1;
#pragma unroll
                for (int off = 16; off >= 1; off >>= 1) ssum += __shfl_xor(ssum, off);
                float inv = 1.f / ssum;
                p0[q] = e0 * inv;
                p1[q] = e1 * inv;
            }
            const _Float16* Vb = Vh + (size_t)b * 1600;
            for (int g = 0; g < 32; ++g) {
                float vr = (float)Vb[(size_t)g * 32 + lane];
#pragma unroll
                for (int q = 0; q < 5; ++q) att[q] += rlane(p0[q], g) * vr;
            }
            for (int g = 32; g < 50; ++g) {
                float vr = (float)Vb[(size_t)g * 32 + lane];
#pragma unroll
                for (int q = 0; q < 5; ++q) att[q] += rlane(p1[q], g - 32) * vr;
            }
        } else {
            const _Float16* Kp = K23 + (size_t)(head - 2) * Bn * 160 + (size_t)b * 160;
            const _Float16* Vp = V23 + (size_t)(head - 2) * Bn * 160 + (size_t)b * 160;
            float k2[5], v2[5];
#pragma unroll
            for (int g = 0; g < 5; ++g) {
                k2[g] = (float)Kp[g * 32 + lane];
                v2[g] = (float)Vp[g * 32 + lane];
            }
#pragma unroll
            for (int q = 0; q < 5; ++q) {
                float s5[5];
#pragma unroll
                for (int g = 0; g < 5; ++g) {
                    float t = qv[q] * k2[g];
#pragma unroll
                    for (int off = 16; off >= 1; off >>= 1) t += __shfl_xor(t, off);
                    s5[g] = t * SCALE;
                }
                float m = s5[0];
#pragma unroll
                for (int g = 1; g < 5; ++g) m = fmaxf(m, s5[g]);
                float e[5], ssum = 0.f;
#pragma unroll
                for (int g = 0; g < 5; ++g) { e[g] = __expf(s5[g] - m); ssum += e[g]; }
                float inv = 1.f / ssum;
                float a = 0.f;
#pragma unroll
                for (int g = 0; g < 5; ++g) a += e[g] * inv * v2[g];
                att[q] = a;
            }
        }
        _Float16* ap = attw + (size_t)head * Bn * 160 + (size_t)b * 160;
#pragma unroll
        for (int q = 0; q < 5; ++q) ap[q * 32 + lane] = (_Float16)att[q];
    }
}

// Cw[i] = Wo[i] @ zipW1[32i:32i+32, :]  (4 x 32x32, f32) -- folds Wo into zip.
__global__ __launch_bounds__(256) void prep_combine_kernel(const float* Wo,
                                                           const float* zipW1,
                                                           float* Cw) {
    for (int e = threadIdx.x; e < 4096; e += 256) {
        int i = e >> 10, k = (e >> 5) & 31, n = e & 31;
        float s = 0.f;
#pragma unroll
        for (int h = 0; h < 32; ++h)
            s += Wo[i * 1024 + k * 32 + h] * zipW1[(i * 32 + h) * 32 + n];
        Cw[e] = s;
    }
}

// z = relu(sum_i att_i @ Cw_i + zipb1) @ zipW2 + zipb2   over rows B*5  (WMMA)
__global__ __launch_bounds__(256) void wo_zip_kernel(
    const _Float16* a0, const _Float16* a1, const _Float16* a2, const _Float16* a3,
    const float* Cw, const float* zipb1, const float* zipW2, const float* zipb2,
    _Float16* z, int R) {
    __shared__ __align__(16) h2 cf[4][2 * 32 * 8];
    __shared__ __align__(16) h2 w2f[2 * 32 * 8];
    __shared__ __align__(16) _Float16 hlds[8][16 * 32];

    int tid = threadIdx.x;
#pragma unroll
    for (int i = 0; i < 4; ++i) build_bfrag(cf[i], Cw + i * 1024, 32, 1, tid, 256);
    build_bfrag(w2f, zipW2, 32, 1, tid, 256);
    __syncthreads();

    int lane = tid & 31, w = tid >> 5;
    int nl = lane & 15, kh = lane >> 4;
    int ntiles = R >> 4;
    int wavesTotal = gridDim.x * 8;
    const _Float16* ap[4] = {a0, a1, a2, a3};

    for (int tile = blockIdx.x * 8 + w; tile < ntiles; tile += wavesTotal) {
        int row0 = tile << 4;
        v8f acc0 = zero8(), acc1 = zero8();
#pragma unroll
        for (int i = 0; i < 4; ++i) {
            v16h a = afrag_g32(ap[i], row0 + nl, kh);
            acc0 = wmma16(a, *(const v16h*)&cf[i][(0 * 32 + lane) * 8], acc0);
            acc1 = wmma16(a, *(const v16h*)&cf[i][(1 * 32 + lane) * 8], acc1);
        }
        float bb0 = zipb1[nl], bb1 = zipb1[nl + 16];
#pragma unroll
        for (int r = 0; r < 8; ++r) {
            acc0[r] = fmaxf(acc0[r] + bb0, 0.f);
            acc1[r] = fmaxf(acc1[r] + bb1, 0.f);
        }
        v16h ah = cd_to_afrag(hlds[w], nl, kh, acc0, acc1);
        v8f o0 = wmma16(ah, *(const v16h*)&w2f[(0 * 32 + lane) * 8], zero8());
        v8f o1 = wmma16(ah, *(const v16h*)&w2f[(1 * 32 + lane) * 8], zero8());
        bb0 = zipb2[nl]; bb1 = zipb2[nl + 16];
#pragma unroll
        for (int r = 0; r < 8; ++r) { o0[r] += bb0; o1[r] += bb1; }
        store_tile_f16(z, row0, nl, kh, o0, o1);
    }
}

// out = relu(z[B,160] @ finW1 + finb1) @ finW2 + finb2   (WMMA, f32 out)
__global__ __launch_bounds__(256) void fin_kernel(
    const _Float16* z, const float* finW1, const float* finb1,
    const float* finW2, const float* finb2, float* out, int R) {
    __shared__ __align__(16) h2 w1f[5 * 2 * 32 * 8];
    __shared__ __align__(16) h2 w2f[2 * 32 * 8];
    __shared__ __align__(16) _Float16 hlds[8][16 * 32];

    int tid = threadIdx.x;
    build_bfrag(w1f, finW1, 160, 5, tid, 256);
    build_bfrag(w2f, finW2, 32, 1, tid, 256);
    __syncthreads();

    int lane = tid & 31, w = tid >> 5;
    int nl = lane & 15, kh = lane >> 4;
    int ntiles = R >> 4;
    int wavesTotal = gridDim.x * 8;

    for (int tile = blockIdx.x * 8 + w; tile < ntiles; tile += wavesTotal) {
        int row0 = tile << 4;
        v8f acc0 = zero8(), acc1 = zero8();
#pragma unroll
        for (int s = 0; s < 5; ++s) {
            // A-frag straight from global f16 [B,160]; 16B aligned
            const _Float16* P = z + (size_t)(row0 + nl) * 160 + s * 32;
            v8h lo = *(const v8h*)(P + kh * 8);
            v8h hi = *(const v8h*)(P + 16 + kh * 8);
            v16h a;
#pragma unroll
            for (int j = 0; j < 8; ++j) { a[j] = lo[j]; a[j + 8] = hi[j]; }
            acc0 = wmma16(a, *(const v16h*)&w1f[((s * 2 + 0) * 32 + lane) * 8], acc0);
            acc1 = wmma16(a, *(const v16h*)&w1f[((s * 2 + 1) * 32 + lane) * 8], acc1);
        }
        float bb0 = finb1[nl], bb1 = finb1[nl + 16];
#pragma unroll
        for (int r = 0; r < 8; ++r) {
            acc0[r] = fmaxf(acc0[r] + bb0, 0.f);
            acc1[r] = fmaxf(acc1[r] + bb1, 0.f);
        }
        v16h ah = cd_to_afrag(hlds[w], nl, kh, acc0, acc1);
        v8f o0 = wmma16(ah, *(const v16h*)&w2f[(0 * 32 + lane) * 8], zero8());
        v8f o1 = wmma16(ah, *(const v16h*)&w2f[(1 * 32 + lane) * 8], zero8());
        bb0 = finb2[nl]; bb1 = finb2[nl + 16];
#pragma unroll
        for (int r = 0; r < 8; ++r) {
            int m = r + 8 * kh;
            out[(size_t)(row0 + m) * 32 + nl]      = o0[r] + bb0;
            out[(size_t)(row0 + m) * 32 + nl + 16] = o1[r] + bb1;
        }
    }
}

extern "C" void kernel_launch(void* const* d_in, const int* in_sizes, int n_in,
                              void* d_out, int out_size, void* d_ws, size_t ws_size,
                              hipStream_t stream) {
    (void)in_sizes; (void)n_in; (void)out_size; (void)ws_size;
    const int B = B_N, NA = NA_N, NC = NC_N, H = H_N;

    const float* agent_x  = (const float*)d_in[0];
    const float* target_x = (const float*)d_in[1];
    const float* cluster_x= (const float*)d_in[2];
    const float* aW1 = (const float*)d_in[3];  const float* ab1 = (const float*)d_in[4];
    const float* aW2 = (const float*)d_in[5];  const float* ab2 = (const float*)d_in[6];
    const float* tW1 = (const float*)d_in[7];  const float* tb1 = (const float*)d_in[8];
    const float* tW2 = (const float*)d_in[9];  const float* tb2 = (const float*)d_in[10];
    const float* cW1 = (const float*)d_in[11]; const float* cb1 = (const float*)d_in[12];
    const float* cW2 = (const float*)d_in[13]; const float* cb2 = (const float*)d_in[14];
    const float* Wq  = (const float*)d_in[15];
    const float* Wk  = (const float*)d_in[16];
    const float* Wv  = (const float*)d_in[17];
    const float* Wo  = (const float*)d_in[18];
    const float* zipW1 = (const float*)d_in[19]; const float* zipb1 = (const float*)d_in[20];
    const float* zipW2 = (const float*)d_in[21]; const float* zipb2 = (const float*)d_in[22];
    const float* finW1 = (const float*)d_in[23]; const float* finb1 = (const float*)d_in[24];
    const float* finW2 = (const float*)d_in[25]; const float* finb2 = (const float*)d_in[26];

    // workspace: K0,V0,K1,V1 [B,50,32]f16; Q(4)/K23(2)/V23(2)/att(4) planes [B,5,32]f16;
    //            Cw 4096 f32; z [B,160]f16.  Total ~139 MB.
    _Float16* ws = (_Float16*)d_ws;
    size_t szKV = (size_t)B * NA * H;
    size_t szQ  = (size_t)B * NC * H;
    _Float16* K0   = ws;
    _Float16* V0   = K0 + szKV;
    _Float16* K1   = V0 + szKV;
    _Float16* V1   = K1 + szKV;
    _Float16* Qw   = V1 + szKV;          // 4 planes
    _Float16* K23  = Qw + 4 * szQ;       // 2 planes
    _Float16* V23  = K23 + 2 * szQ;      // 2 planes
    _Float16* attw = V23 + 2 * szQ;      // 4 planes
    float*    Cw   = (float*)(attw + 4 * szQ);
    _Float16* zbuf = (_Float16*)(Cw + 4096);   // [B,160]

    // ---- agents: encode + K0/V0 ----
    EncArgs ea = {};
    ea.X = agent_x; ea.R = B * NA;
    ea.W1 = aW1; ea.b1 = ab1; ea.W2 = aW2; ea.b2 = ab2;
    ea.PW[0] = Wk + 0 * 1024; ea.Pout[0] = K0;
    ea.PW[1] = Wv + 0 * 1024; ea.Pout[1] = V0;
    enc_proj_kernel<67, 3, 2><<<(B * NA / 16 + 7) / 8, 256, 0, stream>>>(ea);

    // ---- targets: encode + K1/V1 ----
    EncArgs et = {};
    et.X = target_x; et.R = B * NA;
    et.W1 = tW1; et.b1 = tb1; et.W2 = tW2; et.b2 = tb2;
    et.PW[0] = Wk + 1 * 1024; et.Pout[0] = K1;
    et.PW[1] = Wv + 1 * 1024; et.Pout[1] = V1;
    enc_proj_kernel<67, 3, 2><<<(B * NA / 16 + 7) / 8, 256, 0, stream>>>(et);

    // ---- clusters: encode + Q0..Q3, K2/V2, K3/V3 ----
    EncArgs ec = {};
    ec.X = cluster_x; ec.R = B * NC;
    ec.W1 = cW1; ec.b1 = cb1; ec.W2 = cW2; ec.b2 = cb2;
    ec.PW[0] = Wq + 0 * 1024; ec.Pout[0] = Qw + 0 * szQ;
    ec.PW[1] = Wq + 1 * 1024; ec.Pout[1] = Qw + 1 * szQ;
    ec.PW[2] = Wq + 2 * 1024; ec.Pout[2] = Qw + 2 * szQ;
    ec.PW[3] = Wq + 3 * 1024; ec.Pout[3] = Qw + 3 * szQ;
    ec.PW[4] = Wk + 2 * 1024; ec.Pout[4] = K23 + 0 * szQ;
    ec.PW[5] = Wv + 2 * 1024; ec.Pout[5] = V23 + 0 * szQ;
    ec.PW[6] = Wk + 3 * 1024; ec.Pout[6] = K23 + 1 * szQ;
    ec.PW[7] = Wv + 3 * 1024; ec.Pout[7] = V23 + 1 * szQ;
    enc_proj_kernel<105, 4, 8><<<(B * NC / 16 + 7) / 8, 256, 0, stream>>>(ec);

    // ---- fold Wo into zip (tiny) ----
    prep_combine_kernel<<<1, 256, 0, stream>>>(Wo, zipW1, Cw);

    // ---- attention core (softmax part only, VALU) ----
    attn_core_kernel<<<B / 8, 256, 0, stream>>>(K0, V0, K1, V1, Qw, K23, V23, attw, B);

    // ---- Wo+zip (WMMA over B*5 rows) ----
    wo_zip_kernel<<<(B * NC / 16 + 7) / 8, 256, 0, stream>>>(
        attw + 0 * szQ, attw + 1 * szQ, attw + 2 * szQ, attw + 3 * szQ,
        Cw, zipb1, zipW2, zipb2, zbuf, B * NC);

    // ---- fin (WMMA over B rows, f32 out) ----
    fin_kernel<<<(B / 16 + 7) / 8, 256, 0, stream>>>(
        zbuf, finW1, finb1, finW2, finb2, (float*)d_out, B);
}
